// CGRAWindowAttention_61435212202394
// MI455X (gfx1250) — matrix operand order repaired and checked by reading, not compiled, not run on hardware
//
#include <hip/hip_runtime.h>

// ---------------------------------------------------------------------------
// CGRA spiking window attention for gfx1250 (MI455X).
//  - Projection GEMMs:  v_wmma_f32_16x16x4_f32  (4 M-tiles/wave for load amortization)
//  - Attention GEMMs:   v_wmma_f32_16x16x32_f16 (spikes exact in f16)
//  - v-spike tile staged to LDS via TDM tensor_load_to_lds (+ s_wait_tensorcnt)
//  - Deterministic two-stage BN reductions (no float atomics)
// ---------------------------------------------------------------------------

typedef __attribute__((ext_vector_type(2)))  float    f32x2;
typedef __attribute__((ext_vector_type(8)))  float    f32x8;
typedef __attribute__((ext_vector_type(8)))  _Float16 f16x8;
typedef __attribute__((ext_vector_type(16))) _Float16 f16x16;
typedef __attribute__((ext_vector_type(4)))  unsigned int u32x4;
typedef __attribute__((ext_vector_type(8)))  int      i32x8;
typedef __attribute__((ext_vector_type(4)))  int      i32x4;

#define C_DIM   256
#define N_TOK   64
#define H_DIM   8
#define D_HEAD  32
#define BNW     1024
#define NW      64
#define M_ROWS  (BNW * N_TOK)                 // 65536 rows
#define S_ELEMS ((size_t)M_ROWS * C_DIM)      // 16777216 per (BnW,N,C) tensor
#define BN_EPS  1e-5f

#define HAVE_TDM (__has_builtin(__builtin_amdgcn_tensor_load_to_lds) && \
                  __has_builtin(__builtin_amdgcn_s_wait_tensorcnt))

// ---------------------------------------------------------------------------
// K1: feedback  fb[bw,n,c] = bfb[c] + sum_h mean_m(cell[bw,h,n,m]) * Wfb[c,h]
// ---------------------------------------------------------------------------
__global__ __launch_bounds__(256)
void feedback_kernel(const float* __restrict__ cell_mem,
                     const float* __restrict__ Wfb,
                     const float* __restrict__ bfb,
                     float* __restrict__ fb) {
  __shared__ float cs[N_TOK][H_DIM];
  const int bw  = blockIdx.x;
  const int tid = threadIdx.x;
  for (int p = tid; p < H_DIM * N_TOK; p += blockDim.x) {
    const int h = p >> 6, n = p & 63;
    const float* src = cell_mem + (((size_t)bw * H_DIM + h) * N_TOK + n) * N_TOK;
    float s = 0.f;
    for (int m = 0; m < N_TOK; ++m) s += src[m];
    cs[n][h] = s * (1.0f / (float)N_TOK);
  }
  __syncthreads();
  for (int p = tid; p < N_TOK * C_DIM; p += blockDim.x) {
    const int n = p >> 8, c = p & 255;
    float s = bfb[c];
#pragma unroll
    for (int h = 0; h < H_DIM; ++h) s += cs[n][h] * Wfb[c * H_DIM + h];
    fb[(size_t)bw * N_TOK * C_DIM + p] = s;
  }
}

// ---------------------------------------------------------------------------
// K2: fused Q/K/V projection GEMM, y = x @ W^T + b, f32 WMMA 16x16x4.
// One wave computes a 64x16 strip (4 M-tiles) for all three projections:
// per k-step 7 loads feed 12 WMMAs.
// ---------------------------------------------------------------------------
__global__ __launch_bounds__(128)
void qkv_gemm_kernel(const float* __restrict__ x,
                     const float* __restrict__ Wq, const float* __restrict__ bq,
                     const float* __restrict__ Wk, const float* __restrict__ bk,
                     const float* __restrict__ Wv, const float* __restrict__ bv,
                     float* __restrict__ yq, float* __restrict__ yk, float* __restrict__ yv) {
  const int lane = threadIdx.x & 31;
  const int wave = blockIdx.x * (blockDim.x >> 5) + (threadIdx.x >> 5);
  const int ms = wave >> 4;            // 0..1023  (64-row strips)
  const int nt = wave & 15;            // 0..15    (16-col tiles)
  const int ln = lane & 15;
  const int koff = (lane < 16) ? 0 : 2;          // f32 WMMA K split across lane halves
  const int col = nt * 16 + ln;

  const float* xr[4];
#pragma unroll
  for (int t = 0; t < 4; ++t)
    xr[t] = x + (size_t)(ms * 64 + t * 16 + ln) * C_DIM + koff;
  const float* wqr = Wq + (size_t)col * C_DIM + koff;
  const float* wkr = Wk + (size_t)col * C_DIM + koff;
  const float* wvr = Wv + (size_t)col * C_DIM + koff;

  f32x8 cq[4] = {}, ck[4] = {}, cv[4] = {};
#pragma unroll 2
  for (int kk = 0; kk < C_DIM; kk += 4) {
    f32x2 a[4];
#pragma unroll
    for (int t = 0; t < 4; ++t) a[t] = *(const f32x2*)(xr[t] + kk);
    const f32x2 b0 = *(const f32x2*)(wqr + kk);
    const f32x2 b1 = *(const f32x2*)(wkr + kk);
    const f32x2 b2 = *(const f32x2*)(wvr + kk);
#pragma unroll
    for (int t = 0; t < 4; ++t) {
      cq[t] = __builtin_amdgcn_wmma_f32_16x16x4_f32(false, a[t], false, b0, (short)0, cq[t], false, false);
      ck[t] = __builtin_amdgcn_wmma_f32_16x16x4_f32(false, a[t], false, b1, (short)0, ck[t], false, false);
      cv[t] = __builtin_amdgcn_wmma_f32_16x16x4_f32(false, a[t], false, b2, (short)0, cv[t], false, false);
    }
  }
  const float biq = bq[col], bik = bk[col], biv = bv[col];
#pragma unroll
  for (int t = 0; t < 4; ++t) {
    const int mbase = ms * 64 + t * 16 + ((lane < 16) ? 0 : 8);
#pragma unroll
    for (int i = 0; i < 8; ++i) {
      const size_t o = (size_t)(mbase + i) * C_DIM + col;
      yq[o] = cq[t][i] + biq;
      yk[o] = ck[t][i] + bik;
      yv[o] = cv[t][i] + biv;
    }
  }
}

// Output projection: one wave computes 64x32 (4 M-tiles x 2 N-tiles).
__global__ __launch_bounds__(128)
void o_gemm_kernel(const float* __restrict__ a_mat,
                   const float* __restrict__ Wo, const float* __restrict__ bo,
                   float* __restrict__ yo) {
  const int lane = threadIdx.x & 31;
  const int wave = blockIdx.x * (blockDim.x >> 5) + (threadIdx.x >> 5);
  const int ms = wave >> 3;            // 0..1023
  const int np = wave & 7;             // 0..7 (pairs of 16-col tiles)
  const int ln = lane & 15;
  const int koff = (lane < 16) ? 0 : 2;
  const int col0 = np * 32 + ln;
  const int col1 = np * 32 + 16 + ln;

  const float* ar[4];
#pragma unroll
  for (int t = 0; t < 4; ++t)
    ar[t] = a_mat + (size_t)(ms * 64 + t * 16 + ln) * C_DIM + koff;
  const float* wr0 = Wo + (size_t)col0 * C_DIM + koff;
  const float* wr1 = Wo + (size_t)col1 * C_DIM + koff;

  f32x8 c0[4] = {}, c1[4] = {};
#pragma unroll 2
  for (int kk = 0; kk < C_DIM; kk += 4) {
    f32x2 a[4];
#pragma unroll
    for (int t = 0; t < 4; ++t) a[t] = *(const f32x2*)(ar[t] + kk);
    const f32x2 b0 = *(const f32x2*)(wr0 + kk);
    const f32x2 b1 = *(const f32x2*)(wr1 + kk);
#pragma unroll
    for (int t = 0; t < 4; ++t) {
      c0[t] = __builtin_amdgcn_wmma_f32_16x16x4_f32(false, a[t], false, b0, (short)0, c0[t], false, false);
      c1[t] = __builtin_amdgcn_wmma_f32_16x16x4_f32(false, a[t], false, b1, (short)0, c1[t], false, false);
    }
  }
  const float bi0 = bo[col0], bi1 = bo[col1];
#pragma unroll
  for (int t = 0; t < 4; ++t) {
    const int mbase = ms * 64 + t * 16 + ((lane < 16) ? 0 : 8);
#pragma unroll
    for (int i = 0; i < 8; ++i) {
      yo[(size_t)(mbase + i) * C_DIM + col0] = c0[t][i] + bi0;
      yo[(size_t)(mbase + i) * C_DIM + col1] = c1[t][i] + bi1;
    }
  }
}

// ---------------------------------------------------------------------------
// K3: deterministic per-channel stats (two stages, fixed reduction order).
// ---------------------------------------------------------------------------
__global__ __launch_bounds__(256)
void stats_partial_kernel(const float* __restrict__ y, float* __restrict__ partial) {
  const int c = threadIdx.x;
  const int chunk = blockIdx.x;                  // 256 chunks of 256 rows
  const float* p = y + (size_t)chunk * 256 * C_DIM + c;
  float s = 0.f, s2 = 0.f;
  for (int r = 0; r < 256; ++r) {
    const float v = p[(size_t)r * C_DIM];
    s += v; s2 += v * v;
  }
  partial[(size_t)chunk * C_DIM + c] = s;
  partial[(size_t)65536 + (size_t)chunk * C_DIM + c] = s2;
}

__global__ __launch_bounds__(256)
void stats_reduce_kernel(const float* __restrict__ partial, float* __restrict__ stat) {
  const int c = threadIdx.x;
  float s = 0.f, s2 = 0.f;
  for (int b = 0; b < 256; ++b) {
    s  += partial[(size_t)b * C_DIM + c];
    s2 += partial[(size_t)65536 + (size_t)b * C_DIM + c];
  }
  stat[c] = s;
  stat[C_DIM + c] = s2;
}

// ---------------------------------------------------------------------------
// K4: BN(batch stats) + feedback + LIF for q/k/v -> membranes + f16 spikes.
// ---------------------------------------------------------------------------
__global__ __launch_bounds__(256)
void bn_lif_qkv_kernel(const float* __restrict__ yq, const float* __restrict__ yk,
                       const float* __restrict__ yv, const float* __restrict__ fb,
                       const float* __restrict__ stq, const float* __restrict__ stk,
                       const float* __restrict__ stv,
                       const float* __restrict__ gq, const float* __restrict__ btq,
                       const float* __restrict__ gk, const float* __restrict__ btk,
                       const float* __restrict__ gv, const float* __restrict__ btv,
                       const float* __restrict__ qmem, const float* __restrict__ kmem,
                       const float* __restrict__ vmem, const float* __restrict__ betas,
                       float* __restrict__ qm, float* __restrict__ km, float* __restrict__ vm,
                       _Float16* __restrict__ qspk, _Float16* __restrict__ kspk,
                       _Float16* __restrict__ vspk) {
  const size_t idx = (size_t)blockIdx.x * blockDim.x + threadIdx.x;
  if (idx >= S_ELEMS) return;
  const int c = (int)(idx & (C_DIM - 1));
  const float inv = 1.0f / (float)M_ROWS;
  const float f = fb[idx];
  {
    const float m = stq[c] * inv;
    const float var = stq[C_DIM + c] * inv - m * m;
    const float val = gq[c] * (yq[idx] - m) * rsqrtf(var + BN_EPS) + btq[c] + f;
    float mem = betas[0] * qmem[idx] + val;
    const float spk = (mem > 1.0f) ? 1.0f : 0.0f;
    mem -= spk;
    qm[idx] = mem; qspk[idx] = (_Float16)spk;
  }
  {
    const float m = stk[c] * inv;
    const float var = stk[C_DIM + c] * inv - m * m;
    const float val = gk[c] * (yk[idx] - m) * rsqrtf(var + BN_EPS) + btk[c] + f;
    float mem = betas[1] * kmem[idx] + val;
    const float spk = (mem > 1.0f) ? 1.0f : 0.0f;
    mem -= spk;
    km[idx] = mem; kspk[idx] = (_Float16)spk;
  }
  {
    const float m = stv[c] * inv;
    const float var = stv[C_DIM + c] * inv - m * m;
    const float val = gv[c] * (yv[idx] - m) * rsqrtf(var + BN_EPS) + btv[c] + f;
    float mem = betas[2] * vmem[idx] + val;
    const float spk = (mem > 1.0f) ? 1.0f : 0.0f;
    mem -= spk;
    vm[idx] = mem; vspk[idx] = (_Float16)spk;
  }
}

// ---------------------------------------------------------------------------
// TDM: 1-D descriptor load of `nelem` f16 elements into LDS byte offset.
// Group0: count=1, lds_addr, 57-bit global addr, type=2. Group1: data_size=2B,
// tensor_dim0 = tile_dim0 = nelem, stride = nelem. Groups 2/3 unused (zeros).
// ---------------------------------------------------------------------------
__device__ __forceinline__
void tdm_load_f16_1d(unsigned lds_byte_off, const _Float16* gsrc, unsigned nelem) {
#if HAVE_TDM
  const unsigned long long ga = (unsigned long long)(const void*)gsrc;
  u32x4 g0;
  g0[0] = 1u;                                   // count=1 (valid descriptor)
  g0[1] = lds_byte_off;                         // lds_addr
  g0[2] = (unsigned)ga;                         // global_addr[31:0]
  g0[3] = (unsigned)(ga >> 32) | (2u << 30);    // global_addr[56:32] | type=2
  i32x8 g1;
  g1[0] = (int)(1u << 16);                      // data_size = 1 -> 2 bytes
  g1[1] = (int)((nelem & 0xFFFFu) << 16);       // tensor_dim0[15:0]
  g1[2] = (int)((nelem >> 16) & 0xFFFFu);       // tensor_dim0[31:16]
  g1[3] = (int)((nelem & 0xFFFFu) << 16);       // tile_dim0
  g1[4] = 0;                                    // tile_dim1/2 unused
  g1[5] = (int)nelem;                           // tensor_dim0_stride[31:0]
  g1[6] = 0;
  g1[7] = 0;
  const i32x4 z = {0, 0, 0, 0};
#if __clang_major__ >= 23
  const i32x8 z8 = {};
  __builtin_amdgcn_tensor_load_to_lds(g0, g1, z, z, z8, 0);
#else
  __builtin_amdgcn_tensor_load_to_lds(g0, g1, z, z, 0);
#endif
#endif
}

// ---------------------------------------------------------------------------
// K5: attention per window. 1 block per bw, 8 waves = 8 heads.
// v spikes staged LDS via TDM; gate = q k^T * .125 + bias, masked, hard-reset
// cell LIF -> attn spikes (repacked via LDS), out = attn v * .25, attn LIF.
// ---------------------------------------------------------------------------
__global__ __launch_bounds__(256)
void attention_kernel(const _Float16* __restrict__ qspk,
                      const _Float16* __restrict__ kspk,
                      const _Float16* __restrict__ vspk,
                      const float* __restrict__ rel_table,
                      const int*   __restrict__ rel_index,
                      const int*   __restrict__ mask,
                      const float* __restrict__ cell_mem,
                      const float* __restrict__ attn_mem,
                      const float* __restrict__ betas,
                      float* __restrict__ cell_out,
                      float* __restrict__ am_out,
                      float* __restrict__ attnspk) {
  // [0 .. 64KB)  : attn spikes (8 heads x 64 x 64 f16)
  // [64KB..96KB) : staged v spikes (64 x 256 f16) via TDM
  __shared__ _Float16 smem[H_DIM * N_TOK * N_TOK + N_TOK * C_DIM];
  _Float16* attn_s = smem;                         // byte offset 0
  _Float16* v_s    = smem + H_DIM * N_TOK * N_TOK; // byte offset 65536

  const int bw   = blockIdx.x;
  const int h    = threadIdx.x >> 5;               // wave == head
  const int lane = threadIdx.x & 31;
  const int ln   = lane & 15;
  const bool hi  = lane >= 16;
  const float beta_cell = betas[3];
  const float beta_attn = betas[4];
  const int w = bw & (NW - 1);

  // ---- stage v tile (all heads) into LDS: TDM on wave 0, fallback otherwise
#if HAVE_TDM
  if (threadIdx.x < 32) {
    tdm_load_f16_1d(H_DIM * N_TOK * N_TOK * 2u,
                    vspk + (size_t)bw * N_TOK * C_DIM, N_TOK * C_DIM);
    __builtin_amdgcn_s_wait_tensorcnt(0);
  }
#else
  {
    const f16x8* src = (const f16x8*)(vspk + (size_t)bw * N_TOK * C_DIM);
    f16x8* dst = (f16x8*)v_s;
    for (int i = threadIdx.x; i < N_TOK * C_DIM / 8; i += 256) dst[i] = src[i];
  }
#endif

  // ---- load q as A-frags (two 8-half runs / lane), k as B-frags (16-half run)
  f16x16 qA[4], kB[4];
  const _Float16* qbase = qspk + (size_t)bw * N_TOK * C_DIM + h * D_HEAD;
  const _Float16* kbase = kspk + (size_t)bw * N_TOK * C_DIM + h * D_HEAD;
#pragma unroll
  for (int t = 0; t < 4; ++t) {
    const int r = t * 16 + ln;
    const _Float16* pq = qbase + (size_t)r * C_DIM;
    const f16x8 r1 = *(const f16x8*)(pq + (hi ? 16 : 0));
    const f16x8 r2 = *(const f16x8*)(pq + (hi ? 16 : 0) + 8);
    f16x16 a;
#pragma unroll
    for (int j = 0; j < 8; ++j) { a[j] = r1[j]; a[8 + j] = r2[j]; }
    qA[t] = a;
    const _Float16* pk = kbase + (size_t)r * C_DIM;  // r = gate column
    kB[t] = *(const f16x16*)(pk + (hi ? 16 : 0));    // B = k^T by construction
  }

  // ---- gate tiles + bias + mask + hard-reset cell LIF -> attn spikes in LDS
  const float* cellp   = cell_mem + ((size_t)bw * H_DIM + h) * N_TOK * N_TOK;
  float*       cellout = cell_out + ((size_t)bw * H_DIM + h) * N_TOK * N_TOK;
  _Float16*    attn_h  = attn_s + h * N_TOK * N_TOK;
#pragma unroll
  for (int rt = 0; rt < 4; ++rt) {
#pragma unroll
    for (int ct = 0; ct < 4; ++ct) {
      f32x8 acc = {};
      acc = __builtin_amdgcn_wmma_f32_16x16x32_f16(false, qA[rt], false, kB[ct],
                                                   (short)0, acc, false, false);
      const int mcol = ct * 16 + ln;
#pragma unroll
      for (int i = 0; i < 8; ++i) {
        const int nrow = rt * 16 + i + (hi ? 8 : 0);
        float g = acc[i] * 0.125f;
        g += rel_table[rel_index[nrow * N_TOK + mcol] * H_DIM + h];
        if (mask[((size_t)w * N_TOK + nrow) * N_TOK + mcol] != 0) g = 0.0f;
        float mem = beta_cell * cellp[(size_t)nrow * N_TOK + mcol] + g;
        const float spk = (mem > 1.0f) ? 1.0f : 0.0f;
        mem = (spk > 0.0f) ? 0.0f : mem;             // zero (hard) reset
        cellout[(size_t)nrow * N_TOK + mcol] = mem;
        attn_h[nrow * N_TOK + mcol] = (_Float16)spk;
      }
    }
  }
  __syncthreads();   // attn spikes visible + TDM-staged v visible to all waves

  // ---- v as B-frags for out = attn @ v (K = 64 -> two K-steps), from LDS
  f16x16 vB[2][2];
#pragma unroll
  for (int s = 0; s < 2; ++s)
#pragma unroll
    for (int dt = 0; dt < 2; ++dt) {
      f16x16 b;
      const int ks = 32 * s + (hi ? 16 : 0);
      const int ch = h * D_HEAD + dt * 16 + ln;
#pragma unroll
      for (int j = 0; j < 16; ++j)
        b[j] = v_s[(ks + j) * C_DIM + ch];
      vB[s][dt] = b;
    }

  const float* amem = attn_mem + (size_t)bw * N_TOK * C_DIM;
  float*       amo  = am_out   + (size_t)bw * N_TOK * C_DIM;
  float*       aspk = attnspk  + (size_t)bw * N_TOK * C_DIM;
#pragma unroll
  for (int rt = 0; rt < 4; ++rt) {
    f16x16 aA[2];
    const int arow = rt * 16 + ln;
#pragma unroll
    for (int s = 0; s < 2; ++s) {
      const _Float16* lp = attn_h + arow * N_TOK;
      const int off = 32 * s + (hi ? 16 : 0);
      const f16x8 r1 = *(const f16x8*)(lp + off);
      const f16x8 r2 = *(const f16x8*)(lp + off + 8);
      f16x16 a;
#pragma unroll
      for (int j = 0; j < 8; ++j) { a[j] = r1[j]; a[8 + j] = r2[j]; }
      aA[s] = a;
    }
#pragma unroll
    for (int dt = 0; dt < 2; ++dt) {
      f32x8 acc = {};
      acc = __builtin_amdgcn_wmma_f32_16x16x32_f16(false, aA[0], false, vB[0][dt],
                                                   (short)0, acc, false, false);
      acc = __builtin_amdgcn_wmma_f32_16x16x32_f16(false, aA[1], false, vB[1][dt],
                                                   (short)0, acc, false, false);
      const int ch = h * D_HEAD + dt * 16 + ln;
#pragma unroll
      for (int i = 0; i < 8; ++i) {
        const int nrow = rt * 16 + i + (hi ? 8 : 0);
        const float o = acc[i] * 0.25f;
        float mem = beta_attn * amem[(size_t)nrow * C_DIM + ch] + o;
        const float spk = (mem > 0.5f) ? 1.0f : 0.0f;
        mem -= spk * 0.5f;
        amo [(size_t)nrow * C_DIM + ch] = mem;
        aspk[(size_t)nrow * C_DIM + ch] = spk;       // f32 spikes feed O-GEMM
      }
    }
  }
}

// ---------------------------------------------------------------------------
// K8: final BN + LIF (thr 1.0, subtractive reset) -> out spikes and om.
// ---------------------------------------------------------------------------
__global__ __launch_bounds__(256)
void bn_lif_out_kernel(const float* __restrict__ yo, const float* __restrict__ sto,
                       const float* __restrict__ go, const float* __restrict__ bto,
                       const float* __restrict__ omem, const float* __restrict__ betas,
                       float* __restrict__ out_spk, float* __restrict__ om) {
  const size_t idx = (size_t)blockIdx.x * blockDim.x + threadIdx.x;
  if (idx >= S_ELEMS) return;
  const int c = (int)(idx & (C_DIM - 1));
  const float inv = 1.0f / (float)M_ROWS;
  const float m = sto[c] * inv;
  const float var = sto[C_DIM + c] * inv - m * m;
  const float val = go[c] * (yo[idx] - m) * rsqrtf(var + BN_EPS) + bto[c];
  float mem = betas[5] * omem[idx] + val;
  const float spk = (mem > 1.0f) ? 1.0f : 0.0f;
  mem -= spk;
  out_spk[idx] = spk;
  om[idx] = mem;
}

// ---------------------------------------------------------------------------
extern "C" void kernel_launch(void* const* d_in, const int* in_sizes, int n_in,
                              void* d_out, int out_size, void* d_ws, size_t ws_size,
                              hipStream_t stream) {
  (void)in_sizes; (void)n_in; (void)out_size; (void)ws_size;
  const float* x        = (const float*)d_in[0];
  const float* q_mem    = (const float*)d_in[1];
  const float* k_mem    = (const float*)d_in[2];
  const float* v_mem    = (const float*)d_in[3];
  const float* cell_mem = (const float*)d_in[4];
  const float* attn_mem = (const float*)d_in[5];
  const float* out_mem  = (const float*)d_in[6];
  const float* Wq  = (const float*)d_in[7];  const float* bq  = (const float*)d_in[8];
  const float* gq  = (const float*)d_in[9];  const float* btq = (const float*)d_in[10];
  const float* Wk  = (const float*)d_in[11]; const float* bk  = (const float*)d_in[12];
  const float* gk  = (const float*)d_in[13]; const float* btk = (const float*)d_in[14];
  const float* Wv  = (const float*)d_in[15]; const float* bv  = (const float*)d_in[16];
  const float* gv  = (const float*)d_in[17]; const float* btv = (const float*)d_in[18];
  const float* Wfb = (const float*)d_in[19]; const float* bfb = (const float*)d_in[20];
  const float* Wo  = (const float*)d_in[21]; const float* bo  = (const float*)d_in[22];
  const float* go  = (const float*)d_in[23]; const float* bto = (const float*)d_in[24];
  const float* rel_table = (const float*)d_in[25];
  const float* betas     = (const float*)d_in[26];
  const int*   mask      = (const int*)d_in[27];
  const int*   rel_index = (const int*)d_in[28];

  const size_t S = S_ELEMS;
  float* out_base = (float*)d_out;
  float* out_spk = out_base;          // (BnW,N,C)
  float* qm      = out_base + 1 * S;
  float* km      = out_base + 2 * S;
  float* vm      = out_base + 3 * S;
  float* cellout = out_base + 4 * S;  // (BnW*H,N,N) == 2*S elems
  float* am      = out_base + 6 * S;
  float* om      = out_base + 7 * S;

  float* ws = (float*)d_ws;
  float* fbuf = ws;                       // [0,S)
  float* yq   = ws + 1 * S;               // [S,2S)  (reused as yo)
  float* yk   = ws + 2 * S;
  float* yv   = ws + 3 * S;
  _Float16* hb = (_Float16*)(ws + 4 * S); // f16 spike buffers, S halves each
  _Float16* qspk = hb;
  _Float16* kspk = hb + S;
  _Float16* vspk = hb + 2 * S;
  float* attnspk = ws + (11 * S) / 2;     // [5.5S,6.5S) f32 spikes
  float* stats   = ws + (13 * S) / 2;     // 4 x 512 floats
  float* partial = stats + 2048;          // 2 x 65536 floats (reused serially)
  float* yo = yq;                         // safe reuse: yq consumed before K7

  // 1) feedback
  hipLaunchKernelGGL(feedback_kernel, dim3(BNW), dim3(256), 0, stream,
                     cell_mem, Wfb, bfb, fbuf);
  // 2) fused q/k/v projection GEMM (f32 WMMA): 16384 waves
  hipLaunchKernelGGL(qkv_gemm_kernel, dim3(4096), dim3(128), 0, stream,
                     x, Wq, bq, Wk, bk, Wv, bv, yq, yk, yv);
  // 3) deterministic BN stats for q/k/v
  hipLaunchKernelGGL(stats_partial_kernel, dim3(256), dim3(256), 0, stream, yq, partial);
  hipLaunchKernelGGL(stats_reduce_kernel,  dim3(1),   dim3(256), 0, stream, partial, stats + 0);
  hipLaunchKernelGGL(stats_partial_kernel, dim3(256), dim3(256), 0, stream, yk, partial);
  hipLaunchKernelGGL(stats_reduce_kernel,  dim3(1),   dim3(256), 0, stream, partial, stats + 512);
  hipLaunchKernelGGL(stats_partial_kernel, dim3(256), dim3(256), 0, stream, yv, partial);
  hipLaunchKernelGGL(stats_reduce_kernel,  dim3(1),   dim3(256), 0, stream, partial, stats + 1024);
  // 4) BN + feedback + LIF -> membranes + f16 spikes
  hipLaunchKernelGGL(bn_lif_qkv_kernel, dim3(M_ROWS), dim3(256), 0, stream,
                     yq, yk, yv, fbuf, stats + 0, stats + 512, stats + 1024,
                     gq, btq, gk, btk, gv, btv, q_mem, k_mem, v_mem, betas,
                     qm, km, vm, qspk, kspk, vspk);
  // 5) attention (f16 WMMA + TDM v staging), cell LIF, attn LIF
  hipLaunchKernelGGL(attention_kernel, dim3(BNW), dim3(256), 0, stream,
                     qspk, kspk, vspk, rel_table, rel_index, mask,
                     cell_mem, attn_mem, betas, cellout, am, attnspk);
  // 6) output projection GEMM (f32 WMMA): 8192 waves
  hipLaunchKernelGGL(o_gemm_kernel, dim3(2048), dim3(128), 0, stream,
                     attnspk, Wo, bo, yo);
  // 7) BN stats for output
  hipLaunchKernelGGL(stats_partial_kernel, dim3(256), dim3(256), 0, stream, yo, partial);
  hipLaunchKernelGGL(stats_reduce_kernel,  dim3(1),   dim3(256), 0, stream, partial, stats + 1536);
  // 8) final BN + LIF
  hipLaunchKernelGGL(bn_lif_out_kernel, dim3(M_ROWS), dim3(256), 0, stream,
                     yo, stats + 1536, go, bto, out_mem, betas, out_spk, om);
}
